// NnclrVoteQueue_48670569398434
// MI455X (gfx1250) — matrix operand order repaired and checked by reading, not compiled, not run on hardware
//
#include <hip/hip_runtime.h>
#include <math.h>
#include <limits.h>

// ---------------- problem constants ----------------
#define DIMN   256
#define NQ     2
#define NV     2
#define BATCH  512
#define NSIZE  65536
#define VT     16
#define NROWS  (NV * BATCH)          // 1024  (rows of x)
#define QROWS  (NSIZE * NQ)          // 131072 (rows of queue)
#define ROWTILES 64                  // 1024 / 16
#define CHUNKS   64
#define QC       (QROWS / CHUNKS)    // 2048 queue rows per chunk
#define WAVES    8
#define TILES_PER_WAVE (QC / 16 / WAVES)   // 16
#define TSTRIDE  17                  // padded LDS tile row stride (17 coprime 64 banks)

typedef _Float16 v16h __attribute__((ext_vector_type(16)));
typedef _Float16 v8h  __attribute__((ext_vector_type(8)));
typedef float    v8f  __attribute__((ext_vector_type(8)));

union AFrag { v16h v; v8h h[2]; };

// ---------------- top-k helpers (register-resident, fully unrolled) ----------
__device__ __forceinline__ bool kbetter(float av, int ai, float bv, int bi) {
    // descending value, ascending index on ties (jax.lax.top_k semantics)
    return (av > bv) || (av == bv && ai < bi);
}

__device__ __forceinline__ void topk_insert(float (&v)[VT], int (&ix)[VT], float cv, int ci) {
    if (!kbetter(cv, ci, v[VT - 1], ix[VT - 1])) return;
    bool placed = false;
#pragma unroll
    for (int q = VT - 1; q >= 1; --q) {
        bool mv = !placed && kbetter(cv, ci, v[q - 1], ix[q - 1]);
        bool pl = !placed && !mv;
        float nv = mv ? v[q - 1] : (pl ? cv : v[q]);
        int   ni = mv ? ix[q - 1] : (pl ? ci : ix[q]);
        v[q] = nv; ix[q] = ni;
        placed = placed || pl;
    }
    if (!placed) { v[0] = cv; ix[0] = ci; }
}

__device__ __forceinline__ void topk_init(float (&v)[VT], int (&ix)[VT]) {
#pragma unroll
    for (int k = 0; k < VT; ++k) { v[k] = -INFINITY; ix[k] = INT_MAX; }
}

// ---------------- kernel 1: row normalize (f32 src -> f16 normalized [+f32, +invnorm]) ----
__global__ __launch_bounds__(256) void k_norm_rows(const float* __restrict__ src,
                                                   float* __restrict__ dst_f32,
                                                   _Float16* __restrict__ dst_f16,
                                                   float* __restrict__ inv_out,
                                                   int nrows) {
    int row  = blockIdx.x * 8 + (threadIdx.x >> 5);
    int lane = threadIdx.x & 31;
    if (row >= nrows) return;
    const float* r = src + (size_t)row * DIMN;
    float ss = 0.f;
#pragma unroll
    for (int u = 0; u < 8; ++u) { float t = r[lane + 32 * u]; ss += t * t; }
#pragma unroll
    for (int off = 16; off >= 1; off >>= 1) ss += __shfl_xor(ss, off, 32);
    float inv = 1.0f / fmaxf(sqrtf(ss), 1e-12f);
    if (inv_out && lane == 0) inv_out[row] = inv;
#pragma unroll
    for (int u = 0; u < 8; ++u) {
        int d = lane + 32 * u;
        float nv = r[d] * inv;
        if (dst_f32) dst_f32[(size_t)row * DIMN + d] = nv;
        dst_f16[(size_t)row * DIMN + d] = (_Float16)nv;
    }
}

// ---------------- kernel 2: WMMA GEMM + fused streaming per-(row,parity) top-16 ------------
__global__ __launch_bounds__(256) void k_gemm_topk(const _Float16* __restrict__ Xn,
                                                   const _Float16* __restrict__ Qn,
                                                   float* __restrict__ pv,
                                                   int* __restrict__ ps) {
    __shared__ float s_tile[WAVES][16 * TSTRIDE];   // padded 16x16 D transpose buffer per wave
    __shared__ float s_lv[WAVES][32][VT];
    __shared__ int   s_li[WAVES][32][VT];

    const int tid  = threadIdx.x;
    const int wave = tid >> 5;
    const int lane = tid & 31;
    const int rowtile = blockIdx.x;          // 0..63
    const int chunk   = blockIdx.y;          // 0..63
    const int rowbase = rowtile * 16;
    const bool hi = lane >= 16;
    const int  l16 = lane & 15;

    // ---- load A fragments (16 rows x 256 K, f16) once per wave ----
    AFrag a[8];
    const _Float16* xp = Xn + (size_t)(rowbase + l16) * DIMN;
#pragma unroll
    for (int kk = 0; kk < 8; ++kk) {
        int kb = kk * 32;
        a[kk].h[0] = *(const v8h*)(xp + kb + (hi ? 16 : 0));
        a[kk].h[1] = *(const v8h*)(xp + kb + 8 + (hi ? 16 : 0));
    }

    float lv[VT]; int li[VT];
    topk_init(lv, li);

    const int r = l16;            // row within tile owned by this lane for top-k
    const int j = hi ? 1 : 0;     // queue parity owned by this lane

    const int qstart = chunk * QC + wave * TILES_PER_WAVE * 16;
    // lane-fixed base pointer into Qn (column = qbase + l16, k-offset by lane half)
    const _Float16* qpbase = Qn + (size_t)(qstart + l16) * DIMN + (hi ? 16 : 0);

    // ---- software pipeline: double-buffered B fragments across column tiles ----
    v16h bf0[8], bf1[8];
#pragma unroll
    for (int kk = 0; kk < 8; ++kk) bf0[kk] = *(const v16h*)(qpbase + kk * 32);

#pragma unroll 2
    for (int t = 0; t < TILES_PER_WAVE; ++t) {
        v16h (&cur)[8] = (t & 1) ? bf1 : bf0;
        v16h (&nxt)[8] = (t & 1) ? bf0 : bf1;

        // issue next tile's 16 loads now; results consumed only next iteration,
        // so they stay in flight behind the WMMA chain + top-k work below
        if (t + 1 < TILES_PER_WAVE) {
            const _Float16* qp = qpbase + (size_t)(t + 1) * 16 * DIMN;
#pragma unroll
            for (int kk = 0; kk < 8; ++kk) nxt[kk] = *(const v16h*)(qp + kk * 32);
        }

        v8f c = {};
#pragma unroll
        for (int kk = 0; kk < 8; ++kk)
            c = __builtin_amdgcn_wmma_f32_16x16x32_f16(false, a[kk].v, false, cur[kk],
                                                       (short)0, c, false, false);

        // transpose 16x16 D tile through wave-private LDS region (ds ops, padded stride)
#pragma unroll
        for (int i = 0; i < 8; ++i) {
            int m = hi ? (8 + i) : i;
            s_tile[wave][m * TSTRIDE + l16] = c[i];
        }
        // wavefront-scope fence: orders the cross-lane LDS traffic for the compiler,
        // costs no hardware waits (LDS is in-order within a wave)
        __builtin_amdgcn_fence(__ATOMIC_ACQ_REL, "wavefront");

        // gather all 8 transposed values first (one DS clause), then insert
        float vals[8];
#pragma unroll
        for (int u = 0; u < 8; ++u) vals[u] = s_tile[wave][r * TSTRIDE + 2 * u + j];

        int sbase = (qstart + t * 16) >> 1;
#pragma unroll
        for (int u = 0; u < 8; ++u) topk_insert(lv, li, vals[u], sbase + u);

        __builtin_amdgcn_fence(__ATOMIC_ACQ_REL, "wavefront");
    }

#pragma unroll
    for (int k = 0; k < VT; ++k) { s_lv[wave][lane][k] = lv[k]; s_li[wave][lane][k] = li[k]; }
    __syncthreads();

    if (tid < 32) {
        float mv[VT]; int mi[VT];
        topk_init(mv, mi);
        for (int w = 0; w < WAVES; ++w)
#pragma unroll
            for (int k = 0; k < VT; ++k)
                topk_insert(mv, mi, s_lv[w][tid][k], s_li[w][tid][k]);
        size_t base = ((size_t)(rowtile * CHUNKS + chunk) * 32 + tid) * VT;
#pragma unroll
        for (int k = 0; k < VT; ++k) { pv[base + k] = mv[k]; ps[base + k] = mi[k]; }
    }
}

// ---------------- kernel 3: per-b merge + rank voting ------------------------
__global__ __launch_bounds__(128) void k_merge_rank(const float* __restrict__ pv,
                                                    const int* __restrict__ ps,
                                                    int* __restrict__ qidx,
                                                    float* __restrict__ qrank) {
    __shared__ float s_v[4][32][VT];
    __shared__ int   s_i[4][32][VT];
    __shared__ int   s_sel[4][VT];

    const int b = blockIdx.x;
    const int tid = threadIdx.x;
    const int p = tid >> 5;           // 0..3 : p = v*NQ + j
    const int lane = tid & 31;
    const int v = p >> 1, jj = p & 1;
    const int rowtile = v * 32 + (b >> 4);
    const int rr = b & 15;
    const int L2 = jj * 16 + rr;

    float mv[VT]; int mi[VT];
    topk_init(mv, mi);
    for (int t = 0; t < 32; ++t) {
        int e = lane + 32 * t;        // 0..1023 over (chunk,k)
        int c = e >> 4, k = e & 15;
        size_t base = ((size_t)(rowtile * CHUNKS + c) * 32 + L2) * VT + k;
        topk_insert(mv, mi, pv[base], ps[base]);
    }
#pragma unroll
    for (int k = 0; k < VT; ++k) { s_v[p][lane][k] = mv[k]; s_i[p][lane][k] = mi[k]; }
    __syncthreads();

    if (lane == 0) {
        float fv[VT]; int fi[VT];
        topk_init(fv, fi);
        for (int l2 = 0; l2 < 32; ++l2)
#pragma unroll
            for (int k = 0; k < VT; ++k)
                topk_insert(fv, fi, s_v[p][l2][k], s_i[p][l2][k]);
#pragma unroll
        for (int k = 0; k < VT; ++k) s_sel[p][k] = fi[k];
    }
    __syncthreads();

    if (tid == 0) {
        int   sa[64]; float rk[64];
        for (int p2 = 0; p2 < 4; ++p2)
            for (int k = 0; k < VT; ++k) { sa[p2 * VT + k] = s_sel[p2][k]; rk[p2 * VT + k] = (float)k; }
        float sum[64]; bool own[64]; bool taken[64];
        for (int e = 0; e < 64; ++e) {
            float s = 0.f; bool o = true;
            for (int e2 = 0; e2 < 64; ++e2)
                if (sa[e2] == sa[e]) { s += rk[e2]; if (e2 < e) o = false; }
            sum[e] = s; own[e] = o; taken[e] = false;
        }
        int ncur = 0;
        int outi[VT]; float outr[VT];
        for (int t = 0; t < VT; ++t) {
            int best = -1; float bs = INFINITY; int bi = INT_MAX;
            for (int e = 0; e < 64; ++e)
                if (own[e] && !taken[e] && sum[e] > 0.f)
                    if (sum[e] < bs || (sum[e] == bs && sa[e] < bi)) { bs = sum[e]; bi = sa[e]; best = e; }
            if (best >= 0) { taken[best] = true; outi[t] = sa[best]; outr[t] = bs; }
            else {
                for (;;) {
                    bool used = false;
                    for (int tt = 0; tt < t; ++tt) if (outi[tt] == ncur) used = true;
                    if (!used) break;
                    ++ncur;
                }
                outi[t] = ncur; outr[t] = INFINITY; ++ncur;
            }
        }
        for (int t = 0; t < VT; ++t) { qidx[b * VT + t] = outi[t]; qrank[b * VT + t] = outr[t]; }
    }
}

// ---------------- kernel 4: re-score, argmax, copy nn_x, per-b partials -------
__global__ __launch_bounds__(256) void k_finalize(const float* __restrict__ nx,
                                                  const float* __restrict__ qx,
                                                  const int* __restrict__ age,
                                                  const int* __restrict__ qidx,
                                                  const float* __restrict__ qrank,
                                                  const float* __restrict__ invq,
                                                  float* __restrict__ out,
                                                  float* __restrict__ psum) {
    __shared__ float s_sim[32];
    __shared__ int   s_best[2];
    const int b = blockIdx.x;
    const int tid = threadIdx.x;
    const int wave = tid >> 5;
    const int lane = tid & 31;

#pragma unroll
    for (int q = 0; q < 4; ++q) {
        int id = wave * 4 + q;        // 0..31 : (v,k)
        int v = id >> 4, k = id & 15;
        int s = qidx[b * VT + k];
        const float* xr = nx + (size_t)(v * BATCH + b) * DIMN;
        const float* qr = qx + (size_t)s * (NQ * DIMN);   // queue_x[s,0,:]
        float acc = 0.f;
#pragma unroll
        for (int u = 0; u < 8; ++u) { int d = lane + 32 * u; acc += xr[d] * qr[d]; }
#pragma unroll
        for (int off = 16; off >= 1; off >>= 1) acc += __shfl_xor(acc, off, 32);
        if (lane == 0) s_sim[id] = acc * invq[s * NQ];
    }
    __syncthreads();
    if (tid < 2) {
        int v = tid; float best = -INFINITY; int bk = 0;
        for (int k = 0; k < VT; ++k) {
            float sv = s_sim[v * VT + k];
            if (sv > best) { best = sv; bk = k; }   // ties -> lowest k
        }
        s_best[v] = bk;
    }
    __syncthreads();
#pragma unroll
    for (int v = 0; v < 2; ++v) {
        int s = qidx[b * VT + s_best[v]];
        const float* qr = qx + (size_t)s * (NQ * DIMN);
        float* orow = out + (size_t)(v * BATCH + b) * DIMN;
        orow[tid] = qr[tid];
    }
    if (tid == 0) {
        int k0 = s_best[0], k1 = s_best[1];
        int s0 = qidx[b * VT + k0], s1 = qidx[b * VT + k1];
        float simsum = s_sim[k0] + s_sim[16 + k1];
        float agesum = (float)age[s0] + (float)age[s1];
        float rsum = 0.f;
        for (int k = 0; k < VT; ++k) rsum += qrank[b * VT + k];
        psum[b * 3 + 0] = simsum;
        psum[b * 3 + 1] = agesum;
        psum[b * 3 + 2] = rsum;
    }
}

// ---------------- kernel 5: deterministic serial reduction -> means -----------
__global__ void k_means(const float* __restrict__ psum, float* __restrict__ out) {
    if (threadIdx.x == 0 && blockIdx.x == 0) {
        float a = 0.f, g = 0.f, r = 0.f;
        for (int b = 0; b < BATCH; ++b) {
            a += psum[b * 3 + 0]; g += psum[b * 3 + 1]; r += psum[b * 3 + 2];
        }
        out[NROWS * DIMN + 0] = a / (float)NROWS;
        out[NROWS * DIMN + 1] = g / (float)NROWS;
        out[NROWS * DIMN + 2] = r / (float)(BATCH * VT);
    }
}

// ---------------- launcher ----------------------------------------------------
extern "C" void kernel_launch(void* const* d_in, const int* in_sizes, int n_in,
                              void* d_out, int out_size, void* d_ws, size_t ws_size,
                              hipStream_t stream) {
    const float* x   = (const float*)d_in[0];   // (1024, 256)
    const float* qx  = (const float*)d_in[1];   // (65536, 2, 256)
    const int*   age = (const int*)d_in[4];     // (65536,)
    float* out = (float*)d_out;

    char* w = (char*)d_ws;
    size_t off = 0;
    auto alloc = [&](size_t bytes) { char* p = w + off; off = (off + bytes + 255) & ~(size_t)255; return p; };
    _Float16* Xn  = (_Float16*)alloc((size_t)NROWS * DIMN * 2);    // 0.5 MB
    float*    nx  = (float*)  alloc((size_t)NROWS * DIMN * 4);     // 1 MB
    _Float16* Qn  = (_Float16*)alloc((size_t)QROWS * DIMN * 2);    // 64 MB (fits L2)
    float*    invq= (float*)  alloc((size_t)QROWS * 4);            // 0.5 MB
    float*    pv  = (float*)  alloc((size_t)ROWTILES * CHUNKS * 32 * VT * 4);  // 8 MB
    int*      ps  = (int*)    alloc((size_t)ROWTILES * CHUNKS * 32 * VT * 4);  // 8 MB
    int*      qidx= (int*)    alloc((size_t)BATCH * VT * 4);
    float*    qrk = (float*)  alloc((size_t)BATCH * VT * 4);
    float*    psum= (float*)  alloc((size_t)BATCH * 3 * 4);

    // 1) normalize x (f32 + f16) and queue (f16 + inv-norms)
    k_norm_rows<<<NROWS / 8, 256, 0, stream>>>(x, nx, Xn, nullptr, NROWS);
    k_norm_rows<<<QROWS / 8, 256, 0, stream>>>(qx, nullptr, Qn, invq, QROWS);

    // 2) WMMA GEMM fused with streaming per-(row,parity) top-16
    k_gemm_topk<<<dim3(ROWTILES, CHUNKS), 256, 0, stream>>>(Xn, Qn, pv, ps);

    // 3) per-b exact top-16 per view + rank voting
    k_merge_rank<<<BATCH, 128, 0, stream>>>(pv, ps, qidx, qrk);

    // 4) re-score candidates in fp32, pick NN, copy nn_x, per-b partial sums
    k_finalize<<<BATCH, 256, 0, stream>>>(nx, qx, age, qidx, qrk, invq, out, psum);

    // 5) deterministic reduction to the three scalar means
    k_means<<<1, 1, 0, stream>>>(psum, out);
}
